// PooledSelfAttention2d_41540923687268
// MI455X (gfx1250) — matrix operand order repaired and checked by reading, not compiled, not run on hardware
//
#include <hip/hip_runtime.h>
#include <hip/hip_bf16.h>
#include <math.h>

typedef _Float16 half_t;
typedef __attribute__((ext_vector_type(8)))  _Float16 v8h;
typedef __attribute__((ext_vector_type(16))) _Float16 v16h;
typedef __attribute__((ext_vector_type(8)))  float    v8f;

#define Bn 2
#define Cn 64
#define HWn 16384      // 128*128
#define Mn 4096        // HW/4
#define POOLW 64       // pooled width

// ---------------------------------------------------------------------------
// theta = theta_w @ x   -> theta_h[b][n][c8]  (channel-contiguous f16)
// ---------------------------------------------------------------------------
__global__ void __launch_bounds__(256)
proj_theta(const float* __restrict__ x, const float* __restrict__ tw,
           half_t* __restrict__ theta_h) {
    int t = blockIdx.x * 256 + threadIdx.x;        // 0 .. B*HW-1
    int b = t >> 14;
    int n = t & (HWn - 1);
    const float* xp = x + (size_t)b * Cn * HWn + n;
    float acc[8];
    for (int o = 0; o < 8; ++o) acc[o] = 0.f;
    for (int c = 0; c < Cn; ++c) {
        float xv = xp[(size_t)c * HWn];
        for (int o = 0; o < 8; ++o) acc[o] = fmaf(tw[o * Cn + c], xv, acc[o]);
    }
    v8h out;
    for (int o = 0; o < 8; ++o) out[o] = (half_t)acc[o];
    *(v8h*)(theta_h + (size_t)t * 8) = out;
}

// ---------------------------------------------------------------------------
// phi = maxpool2(phi_w @ x) -> phi_h[b][m][c8]  (channel-contiguous f16)
// ---------------------------------------------------------------------------
__global__ void __launch_bounds__(256)
proj_phi(const float* __restrict__ x, const float* __restrict__ pw,
         half_t* __restrict__ phi_h) {
    int t = blockIdx.x * 256 + threadIdx.x;        // 0 .. B*M-1
    int b = t >> 12;
    int m = t & (Mn - 1);
    int i = m >> 6, j = m & (POOLW - 1);
    int p = i * 256 + j * 2;                       // (2i)*128 + 2j
    const float* xp = x + (size_t)b * Cn * HWn;
    float a0[8], a1[8], a2[8], a3[8];
    for (int o = 0; o < 8; ++o) { a0[o] = a1[o] = a2[o] = a3[o] = 0.f; }
    for (int c = 0; c < Cn; ++c) {
        const float* xc = xp + (size_t)c * HWn + p;
        float x0 = xc[0], x1 = xc[1], x2 = xc[128], x3 = xc[129];
        for (int o = 0; o < 8; ++o) {
            float w = pw[o * Cn + c];
            a0[o] = fmaf(w, x0, a0[o]); a1[o] = fmaf(w, x1, a1[o]);
            a2[o] = fmaf(w, x2, a2[o]); a3[o] = fmaf(w, x3, a3[o]);
        }
    }
    v8h out;
    for (int o = 0; o < 8; ++o)
        out[o] = (half_t)fmaxf(fmaxf(a0[o], a1[o]), fmaxf(a2[o], a3[o]));
    *(v8h*)(phi_h + (size_t)t * 8) = out;
}

// ---------------------------------------------------------------------------
// g = maxpool2(g_w @ x) -> g_h[b][c32][m]  (m-contiguous f16)
// ---------------------------------------------------------------------------
__global__ void __launch_bounds__(256)
proj_g(const float* __restrict__ x, const float* __restrict__ gw,
       half_t* __restrict__ g_h) {
    int t = blockIdx.x * 256 + threadIdx.x;        // 0 .. B*32*M-1
    int m  = t & (Mn - 1);
    int oc = (t >> 12) & 31;
    int b  = t >> 17;
    int i = m >> 6, j = m & (POOLW - 1);
    int p = i * 256 + j * 2;
    const float* xp = x + (size_t)b * Cn * HWn;
    const float* w  = gw + oc * Cn;
    float a0 = 0.f, a1 = 0.f, a2 = 0.f, a3 = 0.f;
    for (int c = 0; c < Cn; ++c) {
        const float* xc = xp + (size_t)c * HWn + p;
        float wv = w[c];
        a0 = fmaf(wv, xc[0], a0);   a1 = fmaf(wv, xc[1], a1);
        a2 = fmaf(wv, xc[128], a2); a3 = fmaf(wv, xc[129], a3);
    }
    g_h[t] = (half_t)fmaxf(fmaxf(a0, a1), fmaxf(a2, a3));
}

// ---------------------------------------------------------------------------
// Fused attention: per (b, 16-query tile) keep the full 16x4096 score strip
// in LDS, softmax it there, then value-GEMM from LDS. WMMA f16->f32.
// Block: 128 threads (4 waves). Dynamic LDS: 16*4096 + 512 + 16 + 16 + 128 f32.
// ---------------------------------------------------------------------------
__global__ void __launch_bounds__(128)
attn_kernel(const half_t* __restrict__ theta_h, const half_t* __restrict__ phi_h,
            const half_t* __restrict__ g_h, float* __restrict__ o_pre) {
    extern __shared__ float smem[];
    float* sc     = smem;                 // [16][4096] score strip
    float* Oacc   = smem + 16 * Mn;       // [32][16] output accumulator
    float* rowmax = Oacc + 512;           // [16]
    float* rowsum = rowmax + 16;          // [16]
    float* pred   = rowsum + 16;          // [128] partial reductions

    const int tid   = threadIdx.x;
    const int wave  = tid >> 5;
    const int lane  = tid & 31;
    const int lhalf = lane >> 4;          // 0: lanes 0-15, 1: lanes 16-31
    const int l15   = lane & 15;

    const int blk = blockIdx.x;           // 0..2047
    const int b   = blk >> 10;
    const int n0  = (blk & 1023) << 4;    // query tile start

    // ---- A = theta tile (M=n 16, K=c 8 padded to 32), f16 A layout
    v16h a_th;
    {
        const half_t* tp = theta_h + ((size_t)b * HWn + (n0 + l15)) * 8;
        v8h tv = *(const v8h*)tp;
        for (int i = 0; i < 8; ++i) a_th[i] = lhalf ? (half_t)0.f : tv[i];
        for (int i = 8; i < 16; ++i) a_th[i] = (half_t)0.f;
    }

    // ---- Pass 1: scores -> LDS strip (each wave owns 64 of the 256 m-tiles)
    for (int it = 0; it < 64; ++it) {
        int m0 = (wave * 64 + it) << 4;
        // B = phi (K=c 8 padded to 32, N=m 16), f16 B layout
        v16h bm;
        {
            const half_t* pp = phi_h + ((size_t)b * Mn + (m0 + l15)) * 8;
            v8h pv = *(const v8h*)pp;
            for (int i = 0; i < 8; ++i) bm[i] = lhalf ? (half_t)0.f : pv[i];
            for (int i = 8; i < 16; ++i) bm[i] = (half_t)0.f;
        }
        v8f c0 = {};
        v8f d = __builtin_amdgcn_wmma_f32_16x16x32_f16(
            false, a_th, false, bm, (short)0, c0, false, false);
        // D layout: vgpr r, lanes0-15 -> (row r, col lane); lanes16-31 -> (r+8, lane-16)
        int row0 = lhalf * 8;
        int col  = m0 + l15;
        for (int r = 0; r < 8; ++r) sc[(row0 + r) * Mn + col] = d[r];
    }
    __syncthreads();

    // ---- Pass 2: softmax stats in LDS (row = tid&15, segment = tid>>4)
    {
        int row = tid & 15, seg = tid >> 4;
        const float* p = sc + row * Mn + seg * 512;
        float m = -1e30f;
        for (int i = 0; i < 512; ++i) m = fmaxf(m, p[i]);
        pred[tid] = m;
    }
    __syncthreads();
    if (tid < 16) {
        float m = pred[tid];
        for (int s = 1; s < 8; ++s) m = fmaxf(m, pred[s * 16 + tid]);
        rowmax[tid] = m;
    }
    __syncthreads();
    {
        int row = tid & 15, seg = tid >> 4;
        float rm = rowmax[row];
        float* p = sc + row * Mn + seg * 512;
        float s = 0.f;
        for (int i = 0; i < 512; ++i) { float e = __expf(p[i] - rm); p[i] = e; s += e; }
        pred[tid] = s;
    }
    __syncthreads();
    if (tid < 16) {
        float s = 0.f;
        for (int g = 0; g < 8; ++g) s += pred[g * 16 + tid];
        rowsum[tid] = s;
    }
    for (int i = tid; i < 512; i += 128) Oacc[i] = 0.f;
    __syncthreads();

    // ---- Pass 3: O[c][n] += g[c][m] * P[n][m] over this wave's quarter of m
    v8f d0 = {}, d1 = {};
    const int klo = lhalf * 8;            // A-layout K base for this lane half
    for (int it = 0; it < 32; ++it) {
        int m0 = (wave * 32 + it) << 5;   // 32-wide K chunk
        // A0 = g rows 0..15, A1 = g rows 16..31 (M=c, K=m), f16 A layout
        const half_t* gp0 = g_h + ((size_t)(b * 32) + l15) * Mn + m0;
        const half_t* gp1 = gp0 + 16 * Mn;
        v8h glo0 = *(const v8h*)(gp0 + klo);
        v8h ghi0 = *(const v8h*)(gp0 + 16 + klo);
        v8h glo1 = *(const v8h*)(gp1 + klo);
        v8h ghi1 = *(const v8h*)(gp1 + 16 + klo);
        v16h ag0, ag1;
        for (int i = 0; i < 8; ++i) {
            ag0[i] = glo0[i]; ag0[8 + i] = ghi0[i];
            ag1[i] = glo1[i]; ag1[8 + i] = ghi1[i];
        }
        // B = P^T (K=m 32, N=n 16): lane half selects K range, N = l15
        const float* sp = sc + l15 * Mn + m0 + lhalf * 8;
        v16h bp;
        for (int i = 0; i < 16; ++i) bp[i] = (half_t)sp[(i & 7) + ((i >> 3) << 4)];
        d0 = __builtin_amdgcn_wmma_f32_16x16x32_f16(
            false, ag0, false, bp, (short)0, d0, false, false);
        d1 = __builtin_amdgcn_wmma_f32_16x16x32_f16(
            false, ag1, false, bp, (short)0, d1, false, false);
    }
    // cross-wave reduce into LDS
    {
        int n = l15, c0r = lhalf * 8;
        for (int r = 0; r < 8; ++r) {
            atomicAdd(&Oacc[(c0r + r) * 16 + n], d0[r]);
            atomicAdd(&Oacc[(16 + c0r + r) * 16 + n], d1[r]);
        }
    }
    __syncthreads();
    // normalize by rowsum and write o_pre[b][c][n]
    for (int i = tid; i < 512; i += 128) {
        int c = i >> 4, n = i & 15;
        o_pre[((size_t)b * 32 + c) * HWn + n0 + n] = Oacc[i] / rowsum[n];
    }
}

// ---------------------------------------------------------------------------
// out = gamma * (o_w @ o_pre) + x
// ---------------------------------------------------------------------------
__global__ void __launch_bounds__(256)
out_kernel(const float* __restrict__ o_pre, const float* __restrict__ ow,
           const float* __restrict__ x, const float* __restrict__ gamma,
           float* __restrict__ out) {
    int t = blockIdx.x * 256 + threadIdx.x;   // 0 .. B*64*HW-1
    int n  = t & (HWn - 1);
    int oc = (t >> 14) & 63;
    int b  = t >> 20;
    const float* op = o_pre + (size_t)b * 32 * HWn + n;
    const float* w  = ow + oc * 32;
    float acc = 0.f;
    for (int c = 0; c < 32; ++c) acc = fmaf(w[c], op[(size_t)c * HWn], acc);
    out[t] = gamma[0] * acc + x[t];
}

// ---------------------------------------------------------------------------
extern "C" void kernel_launch(void* const* d_in, const int* in_sizes, int n_in,
                              void* d_out, int out_size, void* d_ws, size_t ws_size,
                              hipStream_t stream) {
    const float* x     = (const float*)d_in[0];
    const float* tw    = (const float*)d_in[1];
    const float* pw    = (const float*)d_in[2];
    const float* gw    = (const float*)d_in[3];
    const float* ow    = (const float*)d_in[4];
    const float* gamma = (const float*)d_in[5];
    float* out = (float*)d_out;

    char* ws = (char*)d_ws;
    half_t* theta_h = (half_t*)ws;                                 // 2*16384*8*2 = 512 KB
    half_t* phi_h   = (half_t*)(ws + 524288);                      // 128 KB
    half_t* g_h     = (half_t*)(ws + 524288 + 131072);             // 512 KB
    float*  o_pre   = (float*)(ws + 524288 + 131072 + 524288);     // 4 MB

    proj_theta<<<(Bn * HWn) / 256, 256, 0, stream>>>(x, tw, theta_h);
    proj_phi  <<<(Bn * Mn) / 256, 256, 0, stream>>>(x, pw, phi_h);
    proj_g    <<<(Bn * 32 * Mn) / 256, 256, 0, stream>>>(x, gw, g_h);

    size_t lds_bytes = (size_t)(16 * Mn + 512 + 16 + 16 + 128) * sizeof(float);
    attn_kernel<<<Bn * (HWn / 16), 128, lds_bytes, stream>>>(theta_h, phi_h, g_h, o_pre);

    out_kernel<<<(Bn * 64 * HWn) / 256, 256, 0, stream>>>(o_pre, ow, x, gamma, out);
}
